// LSTM_52570399703501
// MI455X (gfx1250) — matrix operand order repaired and checked by reading, not compiled, not run on hardware
//
#include <hip/hip_runtime.h>
#include <math.h>

// ---------------------------------------------------------------------------
// Problem constants (from reference): T=128, B=1024, INPUT=4, HID=512, OUT=4,
// FUT=32.  "future" arrives as a device scalar which cannot be read under
// graph capture, so it is fixed here per the reference constants.
// ---------------------------------------------------------------------------
#define TT   128
#define BB   1024
#define HH   512
#define G4H  2048   // 4*HID
#define NOUT 4
#define FUT  32

typedef __attribute__((ext_vector_type(16))) __bf16 v16bf;
typedef __attribute__((ext_vector_type(8)))  float  v8f;
typedef __attribute__((__vector_size__(16))) int    v4i;

// ---------------------------------------------------------------------------
// CDNA5 async global->LDS copy (ASYNCcnt-tracked, bypasses VGPRs) with a
// sync fallback so the file compiles on any toolchain.
// ---------------------------------------------------------------------------
#if __has_builtin(__builtin_amdgcn_global_load_async_to_lds_b128)
#define USE_ASYNC_LDS 1
#else
#define USE_ASYNC_LDS 0
#endif

#if USE_ASYNC_LDS
#if __has_builtin(__builtin_amdgcn_s_wait_asynccnt)
#define WAIT_ASYNC(N) __builtin_amdgcn_s_wait_asynccnt(N)
#else
#define WAIT_ASYNC(N) asm volatile("s_wait_asynccnt %0" ::"n"(N) : "memory")
#endif
#else
#define WAIT_ASYNC(N) ((void)0)
#endif

__device__ __forceinline__ void async_copy16(const void* g, void* l)
{
#if USE_ASYNC_LDS
    // Builtin takes typed v4i pointers: AS(1) global src, AS(3) LDS dst.
    // LDS generic addresses carry the LDS offset in their low 32 bits.
    __builtin_amdgcn_global_load_async_to_lds_b128(
        (__attribute__((address_space(1))) v4i*)(uintptr_t)g,
        (__attribute__((address_space(3))) v4i*)(unsigned)(uintptr_t)l,
        0, 0);
#else
    *(uint4*)l = *(const uint4*)g;
#endif
}

// ---------------------------------------------------------------------------
// Fused LSTM cell:  gate[b, 4H] = A0[b,:] @ W0^T (+ A1[b,:] @ W1^T)
//                                 (+ x[b,0:4] @ Wx^T)  + bih + bhh
//                   c = sig(f)*c_in + sig(i)*tanh(g);  h = sig(o)*tanh(c)
// A*/W* are bf16 (WMMA), x/Wx handled as 4 scalar FMAs in the epilogue.
// Block = 128 threads (4 wave32), each wave owns one gate; block tile is
// 32 (batch) x 32 (hidden).  Grid = (B/32, H/32).  K staged through
// double-buffered LDS via async copies (2-stage software pipeline).
// ---------------------------------------------------------------------------
__global__ __launch_bounds__(128)
void lstm_cell_wmma(const __bf16* __restrict__ A0, const __bf16* __restrict__ W0,
                    const __bf16* __restrict__ A1, const __bf16* __restrict__ W1,
                    const float* __restrict__ bih, const float* __restrict__ bhh,
                    const float* __restrict__ x,   const float* __restrict__ Wx,
                    const float* __restrict__ c_in, float* __restrict__ c_out,
                    float* __restrict__ h_f32, __bf16* __restrict__ h_bf16)
{
    __shared__ alignas(16) __bf16 Ash[2][32][32];     //  4 KB (double buffer)
    __shared__ alignas(16) __bf16 Bsh[2][4][32][32];  // 16 KB (double buffer)
    __shared__ alignas(16) float  Gsh[4][32][32];     // 16 KB

    const int tid  = threadIdx.x;
    const int wave = tid >> 5;         // 0..3  == gate index
    const int lane = tid & 31;
    const int hf   = lane >> 4;        // lane half (0/1)
    const int l16  = lane & 15;

    const int m0 = blockIdx.x * 32;    // batch-row base
    const int n0 = blockIdx.y * 32;    // hidden-col base

    // cooperative 16-byte loader indexing
    const int lr = tid >> 2;           // 0..31 row
    const int lc = (tid & 3) * 8;      // 0,8,16,24 (bf16 elements)

    v8f acc[2][2] = {};                // [m-tile][n-tile] f32 accumulators

    const int nsrc = (A1 != nullptr) ? 2 : 1;
    const int NS   = nsrc * (HH / 32); // pipeline stages

    // issue one stage's global->LDS traffic (5 x b128 per thread)
    auto issue_stage = [&](int buf, int stage) {
        const int srci = stage >> 4;
        const int kk   = (stage & 15) * 32;
        const __bf16* __restrict__ A = srci ? A1 : A0;
        const __bf16* __restrict__ W = srci ? W1 : W0;
        async_copy16(&A[(size_t)(m0 + lr) * HH + kk + lc], &Ash[buf][lr][lc]);
#pragma unroll
        for (int g = 0; g < 4; ++g)
            async_copy16(&W[(size_t)(g * HH + n0 + lr) * HH + kk + lc],
                         &Bsh[buf][g][lr][lc]);
    };

    issue_stage(0, 0);
    for (int s = 0; s < NS; ++s) {
        const int cur = s & 1;
        if (s + 1 < NS) {
            issue_stage(cur ^ 1, s + 1);   // prefetch next slab
            WAIT_ASYNC(5);                 // retire previous stage's 5 ops
        } else {
            WAIT_ASYNC(0);
        }
        __syncthreads();
#pragma unroll
        for (int mi = 0; mi < 2; ++mi) {
            // A fragment (16x32 bf16): lane half 0 -> k 0..7 & 16..23,
            //                          lane half 1 -> k 8..15 & 24..31
            union { v16bf v; uint4 u[2]; } fa;
            const int am = mi * 16 + l16;
            fa.u[0] = *(const uint4*)&Ash[cur][am][hf * 8];
            fa.u[1] = *(const uint4*)&Ash[cur][am][16 + hf * 8];
#pragma unroll
            for (int ni = 0; ni < 2; ++ni) {
                // B fragment (32x16 bf16): col = l16, k = 8*half..+7 & 16+8*half..+7
                union { v16bf v; uint4 u[2]; } fb;
                const int bn = ni * 16 + l16;
                fb.u[0] = *(const uint4*)&Bsh[cur][wave][bn][hf * 8];
                fb.u[1] = *(const uint4*)&Bsh[cur][wave][bn][16 + hf * 8];
                acc[mi][ni] = __builtin_amdgcn_wmma_f32_16x16x32_bf16(
                    false, fa.v, false, fb.v, (short)0, acc[mi][ni],
                    false, false);
            }
        }
        __syncthreads();   // reads of buffer `cur` done before it is rewritten
    }

    // scatter accumulators to LDS gate buffer (D layout: vgpr r -> row r+8*hf)
#pragma unroll
    for (int mi = 0; mi < 2; ++mi)
#pragma unroll
        for (int ni = 0; ni < 2; ++ni)
#pragma unroll
            for (int r = 0; r < 8; ++r)
                Gsh[wave][mi * 16 + r + hf * 8][ni * 16 + l16] = acc[mi][ni][r];
    __syncthreads();

    // epilogue: 1024 elements / 128 threads = 8 each
#pragma unroll
    for (int e = 0; e < 8; ++e) {
        const int idx = tid * 8 + e;
        const int m = idx >> 5;
        const int n = idx & 31;
        const int b    = m0 + m;
        const int hcol = n0 + n;
        float gi = Gsh[0][m][n] + bih[0 * HH + hcol] + bhh[0 * HH + hcol];
        float gf = Gsh[1][m][n] + bih[1 * HH + hcol] + bhh[1 * HH + hcol];
        float gg = Gsh[2][m][n] + bih[2 * HH + hcol] + bhh[2 * HH + hcol];
        float go = Gsh[3][m][n] + bih[3 * HH + hcol] + bhh[3 * HH + hcol];
        if (x != nullptr) {
#pragma unroll
            for (int j = 0; j < 4; ++j) {
                const float xv = x[b * 4 + j];
                gi += xv * Wx[(0 * HH + hcol) * 4 + j];
                gf += xv * Wx[(1 * HH + hcol) * 4 + j];
                gg += xv * Wx[(2 * HH + hcol) * 4 + j];
                go += xv * Wx[(3 * HH + hcol) * 4 + j];
            }
        }
        const float ii = 1.f / (1.f + __expf(-gi));
        const float ff = 1.f / (1.f + __expf(-gf));
        const float gt = tanhf(gg);
        const float oo = 1.f / (1.f + __expf(-go));
        const float c  = ff * c_in[(size_t)b * HH + hcol] + ii * gt;
        const float h  = oo * tanhf(c);
        c_out[(size_t)b * HH + hcol] = c;
        if (h_f32) h_f32[(size_t)b * HH + hcol] = h;
        h_bf16[(size_t)b * HH + hcol] = (__bf16)h;
    }
}

// out[b, o] = h2[b,:] . Wl[o,:] + bl[o]   (4096 dot products of length 512)
__global__ __launch_bounds__(256)
void out_proj(const float* __restrict__ h2, const float* __restrict__ Wl,
              const float* __restrict__ bl, float* __restrict__ out)
{
    const int t = blockIdx.x * 256 + threadIdx.x;   // 0..4095
    const int b = t >> 2;
    const int o = t & 3;
    const float* hr = h2 + (size_t)b * HH;
    const float* wr = Wl + (size_t)o * HH;
    float s = bl[o];
#pragma unroll 4
    for (int k = 0; k < HH; k += 4) {
        s += hr[k] * wr[k] + hr[k + 1] * wr[k + 1] +
             hr[k + 2] * wr[k + 2] + hr[k + 3] * wr[k + 3];
    }
    out[t] = s;
}

__global__ void f32_to_bf16_k(const float* __restrict__ in,
                              __bf16* __restrict__ out, int n)
{
    const int i = blockIdx.x * 256 + threadIdx.x;
    if (i < n) out[i] = (__bf16)in[i];
}

__global__ void zero_u32_k(unsigned int* __restrict__ p, int n)
{
    const int i = blockIdx.x * 256 + threadIdx.x;
    if (i < n) p[i] = 0u;
}

// ---------------------------------------------------------------------------
extern "C" void kernel_launch(void* const* d_in, const int* in_sizes, int n_in,
                              void* d_out, int out_size, void* d_ws, size_t ws_size,
                              hipStream_t stream)
{
    (void)in_sizes; (void)n_in; (void)out_size; (void)ws_size;

    const float* x    = (const float*)d_in[0];   // [T,B,4]
    const float* Wih1 = (const float*)d_in[1];   // [2048,4]
    const float* Whh1 = (const float*)d_in[2];   // [2048,512]
    const float* bih1 = (const float*)d_in[3];
    const float* bhh1 = (const float*)d_in[4];
    const float* Wih2 = (const float*)d_in[5];   // [2048,512]
    const float* Whh2 = (const float*)d_in[6];   // [2048,512]
    const float* bih2 = (const float*)d_in[7];
    const float* bhh2 = (const float*)d_in[8];
    const float* Wl   = (const float*)d_in[9];   // [4,512]
    const float* bl   = (const float*)d_in[10];  // [4]
    float* out = (float*)d_out;                  // [FUT,B,4]

    // -------- workspace layout (bytes) --------
    char* w = (char*)d_ws;
    const size_t W_ELT = (size_t)G4H * HH;       // 1,048,576 elements
    __bf16* Whh1b = (__bf16*)(w);                                  // 2 MB
    __bf16* Wih2b = (__bf16*)(w + 2097152);                        // 2 MB
    __bf16* Whh2b = (__bf16*)(w + 4194304);                        // 2 MB
    __bf16* h1b[2] = { (__bf16*)(w + 6291456), (__bf16*)(w + 7340032) };
    __bf16* h2b[3] = { (__bf16*)(w + 8388608), (__bf16*)(w + 9437184),
                       (__bf16*)(w + 10485760) };
    float*  c1  = (float*)(w + 11534336);                          // 2 MB
    float*  c2  = (float*)(w + 13631488);                          // 2 MB
    float*  h2f = (float*)(w + 15728640);                          // 2 MB

    // -------- one-time per call: convert weights, zero state --------
    {
        const int n = (int)W_ELT;
        const int blocks = (n + 255) / 256;
        f32_to_bf16_k<<<blocks, 256, 0, stream>>>(Whh1, Whh1b, n);
        f32_to_bf16_k<<<blocks, 256, 0, stream>>>(Wih2, Wih2b, n);
        f32_to_bf16_k<<<blocks, 256, 0, stream>>>(Whh2, Whh2b, n);
    }
    // c1+c2 are contiguous: zero 4 MB as one region
    zero_u32_k<<<(1048576 + 255) / 256, 256, 0, stream>>>((unsigned int*)c1, 1048576);
    zero_u32_k<<<(262144 + 255) / 256, 256, 0, stream>>>((unsigned int*)h1b[0], 262144);
    zero_u32_k<<<(262144 + 255) / 256, 256, 0, stream>>>((unsigned int*)h2b[0], 262144);

    const dim3 grid(BB / 32, HH / 32, 1);   // 32 x 16 workgroups
    const dim3 blk(128, 1, 1);

    // -------- phase 1: scan over T timesteps --------
    __bf16* h1_cur = h1b[0]; __bf16* h1_nxt = h1b[1];
    __bf16* h2_cur = h2b[0]; __bf16* h2_nxt = h2b[1];
    for (int t = 0; t < TT; ++t) {
        // cell1: gate = x_t @ Wih1^T + bih1 + h1 @ Whh1^T + bhh1
        lstm_cell_wmma<<<grid, blk, 0, stream>>>(
            h1_cur, Whh1b, nullptr, nullptr, bih1, bhh1,
            x + (size_t)t * BB * 4, Wih1, c1, c1, nullptr, h1_nxt);
        // cell2: gate = h1_new @ Wih2^T + bih2 + h2 @ Whh2^T + bhh2
        lstm_cell_wmma<<<grid, blk, 0, stream>>>(
            h1_nxt, Wih2b, h2_cur, Whh2b, bih2, bhh2,
            nullptr, nullptr, c2, c2, nullptr, h2_nxt);
        __bf16* tswap;
        tswap = h1_cur; h1_cur = h1_nxt; h1_nxt = tswap;
        tswap = h2_cur; h2_cur = h2_nxt; h2_nxt = tswap;
    }
    // after 128 steps h2_cur == h2b[0]; freeze it as tmp_out
    const __bf16* tmp_out = h2_cur;

    // -------- phase 2: FUT future steps (layer-2 weights for both cells) ----
    __bf16* h2w = h2b[2];   // first write target; b0 stays frozen
    for (int ft = 0; ft < FUT; ++ft) {
        lstm_cell_wmma<<<grid, blk, 0, stream>>>(
            tmp_out, Wih2b, h1_cur, Whh2b, bih2, bhh2,
            nullptr, nullptr, c1, c1, nullptr, h1_nxt);
        lstm_cell_wmma<<<grid, blk, 0, stream>>>(
            h1_nxt, Wih2b, h2_cur, Whh2b, bih2, bhh2,
            nullptr, nullptr, c2, c2, h2f, h2w);
        out_proj<<<16, 256, 0, stream>>>(h2f, Wl, bl, out + (size_t)ft * BB * NOUT);
        __bf16* tswap = h1_cur; h1_cur = h1_nxt; h1_nxt = tswap;
        h2_cur = h2w;
        h2w = (h2_cur == h2b[1]) ? h2b[2] : h2b[1];
    }
}
